// MoEMLP_tp_75711683494339
// MI455X (gfx1250) — matrix operand order repaired and checked by reading, not compiled, not run on hardware
//
#include <hip/hip_runtime.h>
#include <hip/hip_bf16.h>

// MoE SequentialMLP: per-expert fc1 -> GELU(tanh) -> fc2, equal capacity.
// E=8, H=2048, F=8192, CAP=2048. Two bf16-WMMA tiled GEMM kernels with a
// bf16 intermediate in d_ws.
//  - fc1: fp32 A/B staged through VGPRs (fp32->bf16 convert), register
//         double-buffered prefetch.
//  - fc2: bf16 A DMA'd global->LDS with GLOBAL_LOAD_ASYNC_TO_LDS_B128
//         (ASYNCcnt), double-buffered LDS; fp32 B via register prefetch.

typedef __bf16 bf16_t;
typedef __attribute__((ext_vector_type(16))) __bf16 v16bf;
typedef __attribute__((ext_vector_type(8)))  float  v8f;

#define NE   8
#define HD   2048
#define FD   8192
#define CAP  2048

#define BM   128
#define BN   128
#define BK   64
#define LDT  72   // LDS row stride (elements); 144B rows keep b128 ops 16B-aligned

__device__ __forceinline__ float gelu_tanh(float x) {
    const float c0 = 0.7978845608028654f;   // sqrt(2/pi)
    const float c1 = 0.044715f;
    float t = tanhf(c0 * (x + c1 * x * x * x));
    return 0.5f * x * (1.0f + t);
}

// ---- WMMA fragment loads from LDS (wave32, 16x16x32 bf16 layouts) ----------
// A (16x32, MxK): lane&15 = row m; lanes>=16 hold K+8:
//   kb = {0,2,4,6}+hi*8 (VGPR0-3) and {16,18,20,22}+hi*8 (VGPR4-7) -> two b128.
__device__ __forceinline__ v16bf load_frag_a(const bf16_t* rowp, int hi) {
    union { uint4 u[2]; v16bf v; } f;
    f.u[0] = *(const uint4*)(rowp + hi * 8);
    f.u[1] = *(const uint4*)(rowp + 16 + hi * 8);
    return f.v;
}
// B (32x16, KxN), stored transposed in LDS as [n][k]: lane&15 = col n,
//   lanes>=16 hold K+8: kb = {0..7}+hi*8 and {16..23}+hi*8 -> two b128.
__device__ __forceinline__ v16bf load_frag_b(const bf16_t* rowp, int hi) {
    union { uint4 u[2]; v16bf v; } f;
    f.u[0] = *(const uint4*)(rowp + hi * 8);
    f.u[1] = *(const uint4*)(rowp + 16 + hi * 8);
    return f.v;
}

__device__ __forceinline__ void cvt4_store(bf16_t* d, float4 v) {
    d[0] = (bf16_t)v.x; d[1] = (bf16_t)v.y; d[2] = (bf16_t)v.z; d[3] = (bf16_t)v.w;
}

#define WMMA_BF16(A, B, C) \
    __builtin_amdgcn_wmma_f32_16x16x32_bf16(false, (A), false, (B), (short)0, (C), false, false)

// ---- gfx1250 async global->LDS (16B per lane, tracked by ASYNCcnt) ---------
__device__ __forceinline__ void async_ld_b128(unsigned lds_byte_off, const void* gaddr) {
    asm volatile("global_load_async_to_lds_b128 %0, %1, off"
                 :: "v"(lds_byte_off),
                    "v"((unsigned long long)(uintptr_t)gaddr)
                 : "memory");
}
__device__ __forceinline__ void wait_async0() {
    asm volatile("s_wait_asynccnt 0" ::: "memory");
}

// ============================ fc1 + GELU =====================================
__global__ __launch_bounds__(256)
void moe_fc1_gelu(const float* __restrict__ x,    // [NE*CAP, HD]
                  const float* __restrict__ w1,   // [NE, HD, FD]
                  const float* __restrict__ b1,   // [NE, FD]
                  bf16_t* __restrict__ hid)       // [NE*CAP, FD]
{
    __shared__ bf16_t As[BM * LDT];
    __shared__ bf16_t Bs[BN * LDT];

    const int e  = blockIdx.z;
    const int m0 = blockIdx.y * BM;
    const int n0 = blockIdx.x * BN;

    const int t    = threadIdx.x;
    const int lane = t & 31;
    const int wid  = t >> 5;
    const int wm   = (wid & 3) * 32;   // 4 waves along M (2 tiles each)
    const int wn   = (wid >> 2) * 64;  // 2 waves along N (4 tiles each)
    const int l15  = lane & 15;
    const int hi   = lane >> 4;

    const float* Ag = x  + (size_t)e * CAP * HD;
    const float* Bg = w1 + (size_t)e * HD * FD;

    v8f acc[2][4] = {};

    const int a_c4 = (t & 15) * 4;   // A: 16 float4 per 64-wide row
    const int a_r0 = t >> 4;
    const int b_c4 = (t & 31) * 4;   // B: 32 float4 per 128-wide row
    const int b_k0 = t >> 5;

    float4 pa[8], pb[8];
#pragma unroll
    for (int i = 0; i < 8; ++i)
        pa[i] = *(const float4*)(Ag + (size_t)(m0 + a_r0 + 16 * i) * HD + a_c4);
#pragma unroll
    for (int i = 0; i < 8; ++i)
        pb[i] = *(const float4*)(Bg + (size_t)(b_k0 + 8 * i) * FD + n0 + b_c4);

    for (int kt = 0; kt < HD; kt += BK) {
        // Commit prefetched tile to LDS (fp32 -> bf16); B stored transposed [n][k]
#pragma unroll
        for (int i = 0; i < 8; ++i)
            cvt4_store(&As[(a_r0 + 16 * i) * LDT + a_c4], pa[i]);
#pragma unroll
        for (int i = 0; i < 8; ++i) {
            int k = b_k0 + 8 * i;
            Bs[(b_c4 + 0) * LDT + k] = (bf16_t)pb[i].x;
            Bs[(b_c4 + 1) * LDT + k] = (bf16_t)pb[i].y;
            Bs[(b_c4 + 2) * LDT + k] = (bf16_t)pb[i].z;
            Bs[(b_c4 + 3) * LDT + k] = (bf16_t)pb[i].w;
        }
        __syncthreads();

        // Prefetch next tile while WMMAs run on this one
        if (kt + BK < HD) {
            const int kn = kt + BK;
#pragma unroll
            for (int i = 0; i < 8; ++i)
                pa[i] = *(const float4*)(Ag + (size_t)(m0 + a_r0 + 16 * i) * HD + kn + a_c4);
#pragma unroll
            for (int i = 0; i < 8; ++i)
                pb[i] = *(const float4*)(Bg + (size_t)(kn + b_k0 + 8 * i) * FD + n0 + b_c4);
        }

#pragma unroll
        for (int ks = 0; ks < BK; ks += 32) {
            v16bf a0 = load_frag_a(&As[(wm +  0 + l15) * LDT + ks], hi);
            v16bf a1 = load_frag_a(&As[(wm + 16 + l15) * LDT + ks], hi);
            v16bf b0 = load_frag_b(&Bs[(wn +  0 + l15) * LDT + ks], hi);
            v16bf b1 = load_frag_b(&Bs[(wn + 16 + l15) * LDT + ks], hi);
            v16bf b2 = load_frag_b(&Bs[(wn + 32 + l15) * LDT + ks], hi);
            v16bf b3 = load_frag_b(&Bs[(wn + 48 + l15) * LDT + ks], hi);
            acc[0][0] = WMMA_BF16(a0, b0, acc[0][0]);
            acc[1][0] = WMMA_BF16(a1, b0, acc[1][0]);
            acc[0][1] = WMMA_BF16(a0, b1, acc[0][1]);
            acc[1][1] = WMMA_BF16(a1, b1, acc[1][1]);
            acc[0][2] = WMMA_BF16(a0, b2, acc[0][2]);
            acc[1][2] = WMMA_BF16(a1, b2, acc[1][2]);
            acc[0][3] = WMMA_BF16(a0, b3, acc[0][3]);
            acc[1][3] = WMMA_BF16(a1, b3, acc[1][3]);
        }
        __syncthreads();
    }

    // Epilogue: bias + GELU, store bf16. C/D layout: n=lane&15, m=v+8*hi.
    const float* bias = b1 + (size_t)e * FD;
#pragma unroll
    for (int tm = 0; tm < 2; ++tm)
#pragma unroll
        for (int tn = 0; tn < 4; ++tn) {
            int gc = n0 + wn + tn * 16 + l15;
            float bz = bias[gc];
#pragma unroll
            for (int v = 0; v < 8; ++v) {
                int gr = m0 + wm + tm * 16 + v + 8 * hi;
                float val = acc[tm][tn][v] + bz;
                hid[((size_t)e * CAP + gr) * FD + gc] = (bf16_t)gelu_tanh(val);
            }
        }
}

// ================================ fc2 ========================================
// Dynamic LDS layout (base offset 0 by construction — no static LDS here):
//   [0, ABYTES)          : A buffer 0
//   [ABYTES, 2*ABYTES)   : A buffer 1
//   [2*ABYTES, +BBYTES)  : B tile
#define ABYTES (BM * LDT * 2)

__global__ __launch_bounds__(256)
void moe_fc2(const bf16_t* __restrict__ hid,   // [NE*CAP, FD]
             const float*  __restrict__ w2,    // [NE, FD, HD]
             const float*  __restrict__ b2,    // [NE, HD]
             float* __restrict__ out)          // [NE*CAP, HD]
{
    extern __shared__ bf16_t smem[];
    bf16_t* Abuf = smem;                    // 2 x [BM*LDT]
    bf16_t* Bs   = smem + 2 * BM * LDT;     // [BN*LDT]

    const int e  = blockIdx.z;
    const int m0 = blockIdx.y * BM;
    const int n0 = blockIdx.x * BN;

    const int t    = threadIdx.x;
    const int lane = t & 31;
    const int wid  = t >> 5;
    const int wm   = (wid & 3) * 32;
    const int wn   = (wid >> 2) * 64;
    const int l15  = lane & 15;
    const int hi   = lane >> 4;

    const bf16_t* Ag = hid + (size_t)e * CAP * FD;
    const float*  Bg = w2  + (size_t)e * FD * HD;

    v8f acc[2][4] = {};

    const int a_c8 = (t & 7) * 8;    // A: 8 x b128(=8 bf16) per 64-wide row
    const int a_r0 = t >> 3;         // 0..31
    const int b_c4 = (t & 31) * 4;
    const int b_k0 = t >> 5;

    // Issue async DMA of A tile kt=0 into buffer 0 (4 x b128 per thread)
#pragma unroll
    for (int i = 0; i < 4; ++i) {
        unsigned l = (unsigned)(((a_r0 + 32 * i) * LDT + a_c8) * 2);
        async_ld_b128(l, Ag + (size_t)(m0 + a_r0 + 32 * i) * FD + a_c8);
    }
    // Prefetch B tile kt=0 into registers
    float4 pb[8];
#pragma unroll
    for (int i = 0; i < 8; ++i)
        pb[i] = *(const float4*)(Bg + (size_t)(b_k0 + 8 * i) * HD + n0 + b_c4);

    int buf = 0;
    for (int kt = 0; kt < FD; kt += BK, buf ^= 1) {
        // Commit B registers to LDS (fp32 -> bf16, transposed [n][k])
#pragma unroll
        for (int i = 0; i < 8; ++i) {
            int k = b_k0 + 8 * i;
            Bs[(b_c4 + 0) * LDT + k] = (bf16_t)pb[i].x;
            Bs[(b_c4 + 1) * LDT + k] = (bf16_t)pb[i].y;
            Bs[(b_c4 + 2) * LDT + k] = (bf16_t)pb[i].z;
            Bs[(b_c4 + 3) * LDT + k] = (bf16_t)pb[i].w;
        }
        wait_async0();          // A tile for this iteration landed in LDS
        __syncthreads();

        // Kick off next tile: A via async DMA into the other buffer,
        // B into registers — both overlap the WMMA burst below.
        if (kt + BK < FD) {
            const int kn = kt + BK;
            const unsigned obase = (unsigned)((buf ^ 1) * ABYTES);
#pragma unroll
            for (int i = 0; i < 4; ++i) {
                unsigned l = obase + (unsigned)(((a_r0 + 32 * i) * LDT + a_c8) * 2);
                async_ld_b128(l, Ag + (size_t)(m0 + a_r0 + 32 * i) * FD + kn + a_c8);
            }
#pragma unroll
            for (int i = 0; i < 8; ++i)
                pb[i] = *(const float4*)(Bg + (size_t)(kn + b_k0 + 8 * i) * HD + n0 + b_c4);
        }

        const bf16_t* As = Abuf + buf * (BM * LDT);
#pragma unroll
        for (int ks = 0; ks < BK; ks += 32) {
            v16bf a0 = load_frag_a(&As[(wm +  0 + l15) * LDT + ks], hi);
            v16bf a1 = load_frag_a(&As[(wm + 16 + l15) * LDT + ks], hi);
            v16bf b0 = load_frag_b(&Bs[(wn +  0 + l15) * LDT + ks], hi);
            v16bf b1 = load_frag_b(&Bs[(wn + 16 + l15) * LDT + ks], hi);
            v16bf b2 = load_frag_b(&Bs[(wn + 32 + l15) * LDT + ks], hi);
            v16bf b3 = load_frag_b(&Bs[(wn + 48 + l15) * LDT + ks], hi);
            acc[0][0] = WMMA_BF16(a0, b0, acc[0][0]);
            acc[1][0] = WMMA_BF16(a1, b0, acc[1][0]);
            acc[0][1] = WMMA_BF16(a0, b1, acc[0][1]);
            acc[1][1] = WMMA_BF16(a1, b1, acc[1][1]);
            acc[0][2] = WMMA_BF16(a0, b2, acc[0][2]);
            acc[1][2] = WMMA_BF16(a1, b2, acc[1][2]);
            acc[0][3] = WMMA_BF16(a0, b3, acc[0][3]);
            acc[1][3] = WMMA_BF16(a1, b3, acc[1][3]);
        }
        __syncthreads();
    }

    const float* bias = b2 + (size_t)e * HD;
#pragma unroll
    for (int tm = 0; tm < 2; ++tm)
#pragma unroll
        for (int tn = 0; tn < 4; ++tn) {
            int gc = n0 + wn + tn * 16 + l15;
            float bz = bias[gc];
#pragma unroll
            for (int v = 0; v < 8; ++v) {
                int gr = m0 + wm + tm * 16 + v + 8 * hi;
                out[((size_t)e * CAP + gr) * HD + gc] = acc[tm][tn][v] + bz;
            }
        }
}

// ============================== launcher =====================================
extern "C" void kernel_launch(void* const* d_in, const int* in_sizes, int n_in,
                              void* d_out, int out_size, void* d_ws, size_t ws_size,
                              hipStream_t stream) {
    (void)in_sizes; (void)n_in; (void)out_size; (void)ws_size;
    const float* x  = (const float*)d_in[0];
    // d_in[1] = tokens_per_expert (equal capacity; unused)
    const float* w1 = (const float*)d_in[2];
    const float* b1 = (const float*)d_in[3];
    const float* w2 = (const float*)d_in[4];
    const float* b2 = (const float*)d_in[5];
    float*  out = (float*)d_out;
    bf16_t* hid = (bf16_t*)d_ws;   // [NE*CAP, FD] bf16 intermediate (256 MiB)

    dim3 blk(256, 1, 1);
    dim3 g1(FD / BN, CAP / BM, NE);   // 64 x 16 x 8
    moe_fc1_gelu<<<g1, blk, 0, stream>>>(x, w1, b1, hid);

    const size_t smem2 = (size_t)(2 * BM * LDT + BN * LDT) * sizeof(bf16_t); // 54 KiB
    dim3 g2(HD / BN, CAP / BM, NE);   // 16 x 16 x 8
    moe_fc2<<<g2, blk, smem2, stream>>>(hid, w2, b2, out);
}
